// STL_47863115546749
// MI455X (gfx1250) — matrix operand order repaired and checked by reading, not compiled, not run on hardware
//
#include <hip/hip_runtime.h>
#include <hip/hip_bf16.h>
#include <math.h>

typedef __attribute__((ext_vector_type(16))) _Float16 v16h;
typedef __attribute__((ext_vector_type(8)))  float    v8f;

#define DIMC  192
#define HEADS 6
#define HDIM  32
#define NTOK  128
#define MLPH  384
#define QK_SCALE 0.17677669529663687f  /* 32^-0.5 */

// ---------------- WMMA fragment helpers (CDNA5 16x16x32 f16 layouts) -------
// A-matrix 16x32 f16: lane l (row = l&15) holds K = 16*(h>>3) + 8*(l>>4) + (h&7)
__device__ __forceinline__ v16h ldA16(const _Float16* p0, int ld, int lane) {
  const _Float16* p = p0 + (lane & 15) * ld + ((lane >> 4) << 3);
  v16h a;
#pragma unroll
  for (int h = 0; h < 8; ++h) a[h] = p[h];
#pragma unroll
  for (int h = 0; h < 8; ++h) a[8 + h] = p[16 + h];
  return a;
}

// B-matrix 32x16 f16 where Bmat[k][n] = W[n0+n][k0+k], W row-major (ld in halfs).
// Lane l (col n = l&15) holds K = 16*(h>>3) + 8*(l>>4) + (h&7).
__device__ __forceinline__ v16h ldB16(const _Float16* p0, int ld, int lane) {
  const _Float16* p = p0 + (lane & 15) * ld + ((lane >> 4) << 3);
  v16h b;
#pragma unroll
  for (int h = 0; h < 16; ++h) b[h] = p[h + (h & 8)];
  return b;
}

__device__ __forceinline__ v8f wmma_f16(v16h a, v16h b, v8f c) {
  return __builtin_amdgcn_wmma_f32_16x16x32_f16(false, a, false, b, (short)0, c,
                                                false, false);
}

// ---------------- kernel 0: weight f32->f16 + rel-pos bias gather ----------
__global__ void prep_kernel(const float* __restrict__ wq, const float* __restrict__ wp,
                            const float* __restrict__ wf1, const float* __restrict__ wf2,
                            const float* __restrict__ rpb, const int* __restrict__ rpi,
                            _Float16* wqH, _Float16* wpH, _Float16* wf1H, _Float16* wf2H,
                            float* biasB) {
  int i = blockIdx.x * blockDim.x + threadIdx.x;
  if (i < 3 * DIMC * DIMC) wqH[i] = (_Float16)wq[i];
  if (i < DIMC * DIMC)     wpH[i] = (_Float16)wp[i];
  if (i < MLPH * DIMC) { wf1H[i] = (_Float16)wf1[i]; wf2H[i] = (_Float16)wf2[i]; }
  if (i < HEADS * NTOK * NTOK) {
    int h = i / (NTOK * NTOK), nm = i % (NTOK * NTOK);
    biasB[i] = rpb[rpi[nm] * HEADS + h];   // bias[h][n][m] = rpb[rpi[n][m]][h]
  }
}

// ---------------- kernel 1: LN1 + roll + window partition -> f16 -----------
// windows (b, zi<4, yi<8, xi<8), token t: dz=t>>6, dy=(t>>3)&7, dx=t&7
// source spatial = ((zi*2+dz+1)&7, (yi*8+dy+4)&63, (xi*8+dx+4)&63)
__global__ void ln_win_kernel(const float* __restrict__ x, const float* __restrict__ g1,
                              const float* __restrict__ b1, _Float16* __restrict__ xw) {
  int win = blockIdx.x, t = threadIdx.x;
  int bb = win >> 8, widx = win & 255;
  int zi = widx >> 6, yi = (widx >> 3) & 7, xi = widx & 7;
  int dz = t >> 6, dy = (t >> 3) & 7, dx = t & 7;
  int d  = (zi * 2 + dz + 1) & 7;
  int hh = (yi * 8 + dy + 4) & 63;
  int ww = (xi * 8 + dx + 4) & 63;
  long sp = ((long)(bb * 8 + d) * 64 + hh) * 64 + ww;
  const float4* r4 = (const float4*)(x + sp * DIMC);
  float s = 0.f, ss = 0.f;
#pragma unroll
  for (int i = 0; i < DIMC / 4; ++i) {
    float4 v = r4[i];
    s  += v.x + v.y + v.z + v.w;
    ss += v.x * v.x + v.y * v.y + v.z * v.z + v.w * v.w;
  }
  float mu = s * (1.0f / DIMC);
  float rinv = rsqrtf(ss * (1.0f / DIMC) - mu * mu + 1e-5f);
  _Float16* o = xw + (long)win * NTOK * DIMC + (long)t * DIMC;
#pragma unroll
  for (int i = 0; i < DIMC / 4; ++i) {
    float4 v = r4[i];
    int c = i * 4;
    o[c + 0] = (_Float16)((v.x - mu) * rinv * g1[c + 0] + b1[c + 0]);
    o[c + 1] = (_Float16)((v.y - mu) * rinv * g1[c + 1] + b1[c + 1]);
    o[c + 2] = (_Float16)((v.z - mu) * rinv * g1[c + 2] + b1[c + 2]);
    o[c + 3] = (_Float16)((v.w - mu) * rinv * g1[c + 3] + b1[c + 3]);
  }
}

// ---------------- kernel 2: per (window, head) fused attention -------------
// LDS: xn 128x192 f16 (49152B, later reused as P 128x128 f16),
//      q 128x32, k 128x32, vT 32x128 (8192B each) => 73728B dynamic
__global__ void __launch_bounds__(256)
attn_kernel(const _Float16* __restrict__ xw, const _Float16* __restrict__ wqH,
            const float* __restrict__ bqkv, const float* __restrict__ biasB,
            const float* __restrict__ maskG, _Float16* __restrict__ attnO) {
  extern __shared__ char smem[];
  _Float16* xnS = (_Float16*)smem;
  _Float16* qS  = (_Float16*)(smem + 49152);
  _Float16* kS  = (_Float16*)(smem + 57344);
  _Float16* vtS = (_Float16*)(smem + 65536);
  int head = blockIdx.x % HEADS;
  int win  = blockIdx.x / HEADS;
  int tid = threadIdx.x, lane = tid & 31, wave = tid >> 5;
  int cl = lane & 15, hi = lane >> 4;

  const _Float16* src = xw + (long)win * NTOK * DIMC;
  for (int i = tid; i < NTOK * DIMC; i += 256) xnS[i] = src[i];
  __syncthreads();

  // ---- QKV for this head (wave w owns rows 16w..16w+15) ----
  v16h a6[6];
#pragma unroll
  for (int kc = 0; kc < 6; ++kc)
    a6[kc] = ldA16(xnS + (wave * 16) * DIMC + kc * 32, DIMC, lane);
  for (int which = 0; which < 3; ++which) {
    int rowbase = which * DIMC + head * HDIM;
#pragma unroll
    for (int nt = 0; nt < 2; ++nt) {
      v8f acc = {};
#pragma unroll
      for (int kc = 0; kc < 6; ++kc)
        acc = wmma_f16(a6[kc],
                       ldB16(wqH + (long)(rowbase + nt * 16) * DIMC + kc * 32, DIMC, lane),
                       acc);
      int dcol = nt * 16 + cl;
      float bias = bqkv[rowbase + dcol];
#pragma unroll
      for (int r = 0; r < 8; ++r) {
        int n = wave * 16 + r + hi * 8;
        float v = acc[r] + bias;
        if (which == 0)      qS[n * HDIM + dcol]  = (_Float16)(v * QK_SCALE);
        else if (which == 1) kS[n * HDIM + dcol]  = (_Float16)v;
        else                 vtS[dcol * NTOK + n] = (_Float16)v;  // V transposed
      }
    }
  }
  __syncthreads();

  // ---- S = q k^T : one WMMA per 16x16 tile (K = HDIM = 32) ----
  v16h aq = ldA16(qS + wave * 16 * HDIM, HDIM, lane);
  v8f s[8];
#pragma unroll
  for (int j = 0; j < 8; ++j) {
    v8f z = {};
    s[j] = wmma_f16(aq, ldB16(kS + j * 16 * HDIM, HDIM, lane), z);
  }
  // ---- + rel-pos bias + shift mask ----
  int widx = win & 255;
  const float* bh = biasB + (long)head * NTOK * NTOK;
  const float* mk = maskG + (long)widx * NTOK * NTOK;
  int rowoff = wave * 16 + hi * 8;
#pragma unroll
  for (int j = 0; j < 8; ++j)
#pragma unroll
    for (int r = 0; r < 8; ++r) {
      int n = rowoff + r, m = j * 16 + cl;
      s[j][r] += bh[n * NTOK + m] + mk[n * NTOK + m];
    }
  // ---- softmax per row (8 regs x 16 lanes hold one row) ----
  _Float16* pS = (_Float16*)smem;  // reuse xn region (no reader remains)
#pragma unroll
  for (int r = 0; r < 8; ++r) {
    float mx = -1e30f;
#pragma unroll
    for (int j = 0; j < 8; ++j) mx = fmaxf(mx, s[j][r]);
#pragma unroll
    for (int off = 1; off < 16; off <<= 1) mx = fmaxf(mx, __shfl_xor(mx, off, 32));
    float sum = 0.f;
#pragma unroll
    for (int j = 0; j < 8; ++j) { float e = __expf(s[j][r] - mx); s[j][r] = e; sum += e; }
#pragma unroll
    for (int off = 1; off < 16; off <<= 1) sum += __shfl_xor(sum, off, 32);
    float rs = 1.f / sum;
    int n = rowoff + r;
#pragma unroll
    for (int j = 0; j < 8; ++j) pS[n * NTOK + j * 16 + cl] = (_Float16)(s[j][r] * rs);
  }
  // ---- out = P V (each wave reads only its own P rows; vT synced above) ----
#pragma unroll
  for (int j2 = 0; j2 < 2; ++j2) {
    v8f acc = {};
#pragma unroll
    for (int kc = 0; kc < 4; ++kc)
      acc = wmma_f16(ldA16(pS + wave * 16 * NTOK + kc * 32, NTOK, lane),
                     ldB16(vtS + j2 * 16 * NTOK + kc * 32, NTOK, lane), acc);
    int dcol = j2 * 16 + cl;
#pragma unroll
    for (int r = 0; r < 8; ++r) {
      int n = wave * 16 + r + hi * 8;
      attnO[(long)win * NTOK * DIMC + (long)n * DIMC + head * HDIM + dcol] = (_Float16)acc[r];
    }
  }
}

// ---------------- kernel 3: proj + residual + window reverse ---------------
__global__ void __launch_bounds__(256)
proj_kernel(const _Float16* __restrict__ attnI, const _Float16* __restrict__ wpH,
            const float* __restrict__ bproj, const float* __restrict__ x,
            float* __restrict__ x1) {
  extern __shared__ char smem[];
  _Float16* aS = (_Float16*)smem;  // 128x192 f16
  int win = blockIdx.x, tid = threadIdx.x, lane = tid & 31, wave = tid >> 5;
  int cl = lane & 15, hi = lane >> 4;
  const _Float16* src = attnI + (long)win * NTOK * DIMC;
  for (int i = tid; i < NTOK * DIMC; i += 256) aS[i] = src[i];
  __syncthreads();
  int bb = win >> 8, widx = win & 255;
  int zi = widx >> 6, yi = (widx >> 3) & 7, xi = widx & 7;
  long smap[8];
#pragma unroll
  for (int r = 0; r < 8; ++r) {
    int n = wave * 16 + r + hi * 8;
    int dz = n >> 6, dy = (n >> 3) & 7, dx = n & 7;
    int d  = (zi * 2 + dz + 1) & 7;
    int hh = (yi * 8 + dy + 4) & 63;
    int ww = (xi * 8 + dx + 4) & 63;
    smap[r] = ((long)(bb * 8 + d) * 64 + hh) * 64 + ww;
  }
  v16h a6[6];
#pragma unroll
  for (int kc = 0; kc < 6; ++kc)
    a6[kc] = ldA16(aS + wave * 16 * DIMC + kc * 32, DIMC, lane);
#pragma unroll
  for (int j = 0; j < 12; ++j) {
    v8f acc = {};
#pragma unroll
    for (int kc = 0; kc < 6; ++kc)
      acc = wmma_f16(a6[kc], ldB16(wpH + (long)(j * 16) * DIMC + kc * 32, DIMC, lane), acc);
    int c = j * 16 + cl;
    float bp = bproj[c];
#pragma unroll
    for (int r = 0; r < 8; ++r) {
      long sp = smap[r];
      x1[sp * DIMC + c] = acc[r] + bp + x[sp * DIMC + c];
    }
  }
}

// ---------------- kernel 4: LN2 + MLP (exact GELU) + residual --------------
__global__ void __launch_bounds__(256)
mlp_kernel(const float* __restrict__ x1, const float* __restrict__ g2,
           const float* __restrict__ b2, const _Float16* __restrict__ wf1H,
           const float* __restrict__ bf1, const _Float16* __restrict__ wf2H,
           const float* __restrict__ bf2, float* __restrict__ out) {
  extern __shared__ char smem[];
  _Float16* zS = (_Float16*)smem;            // 128x192 f16
  _Float16* hS = (_Float16*)(smem + 49152);  // 128x384 f16
  int tid = threadIdx.x, lane = tid & 31, wave = tid >> 5;
  int cl = lane & 15, hi = lane >> 4;
  long tb = (long)blockIdx.x * NTOK;
  int rowoff = wave * 16 + hi * 8;
  // LN2 (row lives across 12 regs x 16 lanes)
#pragma unroll
  for (int r = 0; r < 8; ++r) {
    long n = tb + rowoff + r;
    float v[12];
    float s = 0.f;
#pragma unroll
    for (int j = 0; j < 12; ++j) { v[j] = x1[n * DIMC + j * 16 + cl]; s += v[j]; }
#pragma unroll
    for (int off = 1; off < 16; off <<= 1) s += __shfl_xor(s, off, 32);
    float mu = s * (1.0f / DIMC);
    float ss = 0.f;
#pragma unroll
    for (int j = 0; j < 12; ++j) { float d = v[j] - mu; ss += d * d; }
#pragma unroll
    for (int off = 1; off < 16; off <<= 1) ss += __shfl_xor(ss, off, 32);
    float rinv = rsqrtf(ss * (1.0f / DIMC) + 1e-5f);
#pragma unroll
    for (int j = 0; j < 12; ++j) {
      int c = j * 16 + cl;
      zS[(rowoff + r) * DIMC + c] = (_Float16)((v[j] - mu) * rinv * g2[c] + b2[c]);
    }
  }
  __syncthreads();
  // fc1 + exact GELU
  v16h a6[6];
#pragma unroll
  for (int kc = 0; kc < 6; ++kc)
    a6[kc] = ldA16(zS + wave * 16 * DIMC + kc * 32, DIMC, lane);
  for (int nt = 0; nt < 24; ++nt) {
    v8f acc = {};
#pragma unroll
    for (int kc = 0; kc < 6; ++kc)
      acc = wmma_f16(a6[kc], ldB16(wf1H + (long)(nt * 16) * DIMC + kc * 32, DIMC, lane), acc);
    int c = nt * 16 + cl;
    float bias = bf1[c];
#pragma unroll
    for (int r = 0; r < 8; ++r) {
      float u = acc[r] + bias;
      float g = 0.5f * u * (1.f + erff(u * 0.70710678118654752f));
      hS[(rowoff + r) * MLPH + c] = (_Float16)g;
    }
  }
  __syncthreads();
  // fc2 + residual
  v16h a12[12];
#pragma unroll
  for (int kc = 0; kc < 12; ++kc)
    a12[kc] = ldA16(hS + wave * 16 * MLPH + kc * 32, MLPH, lane);
#pragma unroll
  for (int nt = 0; nt < 12; ++nt) {
    v8f acc = {};
#pragma unroll
    for (int kc = 0; kc < 12; ++kc)
      acc = wmma_f16(a12[kc], ldB16(wf2H + (long)(nt * 16) * MLPH + kc * 32, MLPH, lane), acc);
    int c = nt * 16 + cl;
    float bias = bf2[c];
#pragma unroll
    for (int r = 0; r < 8; ++r) {
      long n = tb + rowoff + r;
      out[n * DIMC + c] = acc[r] + bias + x1[n * DIMC + c];
    }
  }
}

// ---------------- launcher --------------------------------------------------
extern "C" void kernel_launch(void* const* d_in, const int* in_sizes, int n_in,
                              void* d_out, int out_size, void* d_ws, size_t ws_size,
                              hipStream_t stream) {
  const float* x     = (const float*)d_in[0];
  const float* maskM = (const float*)d_in[1];
  const int*   rpi   = (const int*)d_in[2];
  const float* g1    = (const float*)d_in[3];
  const float* b1    = (const float*)d_in[4];
  const float* wqkv  = (const float*)d_in[5];
  const float* bqkv  = (const float*)d_in[6];
  const float* rpb   = (const float*)d_in[7];
  const float* wproj = (const float*)d_in[8];
  const float* bproj = (const float*)d_in[9];
  const float* g2    = (const float*)d_in[10];
  const float* b2    = (const float*)d_in[11];
  const float* wfc1  = (const float*)d_in[12];
  const float* bfc1  = (const float*)d_in[13];
  const float* wfc2  = (const float*)d_in[14];
  const float* bfc2  = (const float*)d_in[15];
  float* out = (float*)d_out;
  char* ws = (char*)d_ws;

  _Float16* wqH   = (_Float16*)(ws + 0);          // 576*192 f16
  _Float16* wpH   = (_Float16*)(ws + 221184);     // 192*192 f16
  _Float16* wf1H  = (_Float16*)(ws + 294912);     // 384*192 f16
  _Float16* wf2H  = (_Float16*)(ws + 442368);     // 192*384 f16
  float*    biasB = (float*)   (ws + 589824);     // 6*128*128 f32
  _Float16* xw    = (_Float16*)(ws + 983040);     // 512*128*192 f16
  _Float16* attn  = (_Float16*)(ws + 26148864);   // 512*128*192 f16
  float*    x1    = (float*)   (ws + 51314688);   // 65536*192 f32

  prep_kernel<<<432, 256, 0, stream>>>(wqkv, wproj, wfc1, wfc2, rpb, rpi,
                                       wqH, wpH, wf1H, wf2H, biasB);
  ln_win_kernel<<<512, 128, 0, stream>>>(x, g1, b1, xw);
  attn_kernel<<<512 * HEADS, 256, 73728, stream>>>(xw, wqH, bqkv, biasB, maskM, attn);
  proj_kernel<<<512, 256, 49152, stream>>>(attn, wpH, bproj, x, x1);
  mlp_kernel<<<512, 256, 147456, stream>>>(x1, g2, b2, wf1H, bfc1, wf2H, bfc2, out);
}
